// GraniteSpeechConformerAttention_4973572129233
// MI455X (gfx1250) — matrix operand order repaired and checked by reading, not compiled, not run on hardware
//
#include <hip/hip_runtime.h>
#include <hip/hip_bf16.h>

// ---------------- problem constants (from reference) ----------------
#define B_    16
#define N_    3900
#define D_    512
#define C_    200      // context_size
#define NB_   20       // ceil(3900/200)
#define NR_   100      // 3900 % 200
#define NP_   4000     // padded tokens per batch
#define H_    8
#define DH_   64
#define CP_   224      // keys padded to 14 WMMA tiles (mult of 32 for P@V k-loop)
#define TSR_  416      // rel-pos band rows staged (399 valid, mult of 16)

// LDS strides chosen for conflict-free access on 64 x 4B banks
#define KS_ST 72       // K / rel-table row stride (halves): 36 dw -> distinct banks
#define VT_ST 232      // transposed-V / P row stride (halves): 116 dw, gcd(52,64)=4
#define S_ST  225      // score row stride (floats): 225 %% 64 = 33, odd-ish
#define XS_ST 520      // GEMM A-tile row stride (halves): 260 dw, step 4 banks

typedef __attribute__((ext_vector_type(16))) _Float16 v16h;
typedef __attribute__((ext_vector_type(8)))  _Float16 h8;
typedef __attribute__((ext_vector_type(8)))  float    v8f;

#define WMMA_F16(a, b, c) \
  __builtin_amdgcn_wmma_f32_16x16x32_f16(false, (a), false, (b), (short)0, (c), false, false)

// gfx1250 async global->LDS 16B copy (tracked by ASYNCcnt, not LOADcnt)
__device__ __forceinline__ void async_copy_b128(unsigned lds_byte_addr,
                                                const void* gsrc) {
  asm volatile("global_load_async_to_lds_b128 %0, %1, off"
               :: "v"(lds_byte_addr), "v"((unsigned long long)gsrc)
               : "memory");
}
__device__ __forceinline__ void async_wait0() {
  asm volatile("s_wait_asynccnt 0x0" ::: "memory");
}
// low 32 bits of a generic pointer into LDS == wave-relative LDS byte offset
__device__ __forceinline__ unsigned lds_addr_of(const void* p) {
  return (unsigned)(unsigned long long)p;
}

// build a 16-half A/B fragment from two contiguous 16B chunks
__device__ __forceinline__ v16h frag2(const _Float16* p0, const _Float16* p1) {
  h8 lo = *(const h8*)p0;
  h8 hi = *(const h8*)p1;
  return __builtin_shufflevector(lo, hi, 0, 1, 2, 3, 4, 5, 6, 7,
                                 8, 9, 10, 11, 12, 13, 14, 15);
}

// =====================================================================
// Kernel 0: pack fp32 weight [512, ncols] into WMMA-B fragment order:
//   Wp[((nt*16 + kt)*32 + lane)*16 + e] = W[(kt*32 + 8*(lane>>4) + (e&7) + 16*(e>>3))*ncols
//                                           + nt*16 + (lane&15)]
// so each lane's 16 B elements for one (nt, k-step) are one 32B chunk.
// =====================================================================
__global__ void pack_w_kernel(const float* __restrict__ W, _Float16* __restrict__ Wp,
                              int ncols) {
  int i = blockIdx.x * blockDim.x + threadIdx.x;
  if (i >= 512 * ncols) return;
  int e = i & 15;
  int lane = (i >> 4) & 31;
  int kt = (i >> 9) & 15;
  int nt = i >> 13;
  int k = kt * 32 + ((lane >> 4) << 3) + (e & 7) + ((e >> 3) << 4);
  int col = nt * 16 + (lane & 15);
  Wp[i] = (_Float16)W[(size_t)k * ncols + col];
}

// =====================================================================
// Kernel 1: Y[b,t,:] = f16( x_pad[b,t,:] @ W )   (A fp32, staged in LDS)
// grid.x = 4000 M-tiles, grid.y * 8 waves = N tiles. block = 256 (8 waves)
// =====================================================================
__global__ void __launch_bounds__(256)
gemm_qkv_kernel(const float* __restrict__ X, const _Float16* __restrict__ Wp,
                _Float16* __restrict__ Y, int ncols) {
  __shared__ _Float16 Xs[16 * XS_ST];
  const int tid = threadIdx.x;
  const int mt = blockIdx.x;
  // cooperative coalesced stage of 16x512 A tile (fp32 -> f16), zero padded rows
  for (int idx = tid; idx < 16 * 512; idx += 256) {
    int r = idx >> 9, k = idx & 511;
    int arow = mt * 16 + r;
    int bb = arow / NP_, tt = arow - bb * NP_;
    float v = (tt < N_) ? X[((size_t)bb * N_ + tt) * D_ + k] : 0.f;
    Xs[r * XS_ST + k] = (_Float16)v;
  }
  __syncthreads();

  const int lane = tid & 31, wid = tid >> 5;
  const int col16 = lane & 15, hf = lane >> 4;
  const int nt = blockIdx.y * 8 + wid;
  const int col = nt * 16 + col16;
  const _Float16* xrow = Xs + col16 * XS_ST;
  v8f acc = {};
#pragma unroll 2
  for (int k0 = 0; k0 < 512; k0 += 32) {
    v16h a = frag2(xrow + k0 + 8 * hf, xrow + k0 + 16 + 8 * hf);
    v16h bf = *(const v16h*)(Wp + (((size_t)nt * 16 + (k0 >> 5)) * 32 + lane) * 16);
    acc = WMMA_F16(a, bf, acc);
  }
#pragma unroll
  for (int r = 0; r < 8; ++r) {
    int mrow = mt * 16 + r + 8 * hf;
    Y[(size_t)mrow * ncols + col] = (_Float16)acc[r];
  }
}

// =====================================================================
// Kernel 2: fused chunk attention. One block per (b, chunk, head).
// block = 128 (4 waves); each wave owns q-tiles {w, w+4, w+8, w+12} (<13).
// K rows staged with async global->LDS b128 copies (ASYNCcnt); V staged
// transposed and the rel-pos table converted fp32->f16 on the regular path.
// LDS ~204 KB dynamic (< 320 KB/WGP). ctx output aliases q16 (disjoint
// read/write regions per block).
// =====================================================================
__global__ void __launch_bounds__(128)
attn_kernel(const _Float16* __restrict__ q16, const _Float16* __restrict__ kv16,
            const float* __restrict__ relT, _Float16* __restrict__ ctx) {
  extern __shared__ __attribute__((aligned(32))) char smem_raw[];
  _Float16* Ks = reinterpret_cast<_Float16*>(smem_raw);        // [CP_][KS_ST]
  _Float16* Vt = Ks + CP_ * KS_ST;                             // [DH_][VT_ST] transposed
  _Float16* Ts = Vt + DH_ * VT_ST;                             // [TSR_][KS_ST]
  float* Sw = reinterpret_cast<float*>(Ts + TSR_ * KS_ST);     // [4][16][S_ST]
  _Float16* Pw = reinterpret_cast<_Float16*>(Sw + 4 * 16 * S_ST);  // [4][16][VT_ST]
  float* sInv = reinterpret_cast<float*>(Pw + 4 * 16 * VT_ST);     // [4][16]

  const int m = blockIdx.x / H_;
  const int h = blockIdx.x % H_;
  const int b = blockIdx.y;
  const int tid = threadIdx.x;
  const int lane = tid & 31, wid = tid >> 5;
  const int col16 = lane & 15, hf = lane >> 4;
  const bool last = (m == NB_ - 1);

  // prefetch the Q rows this block will consume (global_prefetch)
  for (int r = tid; r < C_; r += 128)
    __builtin_prefetch(q16 + ((size_t)(b * NP_ + m * C_ + r)) * 512 + h * DH_, 0, 1);

  // ---- stage K rows via async global->LDS (16B per lane-chunk) ----
  {
    const unsigned ldsK = lds_addr_of(Ks);
    const _Float16* ksrc = kv16 + ((size_t)(b * NP_ + m * C_)) * 1024 + h * DH_;
    for (int idx = tid; idx < C_ * 8; idx += 128) {       // 200 rows x 8 chunks
      int r = idx >> 3, c8 = (idx & 7) * 8;
      async_copy_b128(ldsK + (unsigned)(r * KS_ST + c8) * 2,
                      ksrc + (size_t)r * 1024 + c8);
    }
    for (int idx = tid; idx < (CP_ - C_) * DH_; idx += 128) {  // zero pad rows
      int r = C_ + (idx >> 6), c = idx & 63;
      Ks[r * KS_ST + c] = (_Float16)0;
    }
  }
  // ---- stage transposed V (f16, zero pad keys >= 200) ----
  for (int idx = tid; idx < CP_ * DH_; idx += 128) {
    int r = idx >> 6, c = idx & 63;
    _Float16 vv = (_Float16)0;
    if (r < C_)
      vv = kv16[((size_t)(b * NP_ + m * C_ + r)) * 1024 + 512 + h * DH_ + c];
    Vt[c * VT_ST + r] = vv;
  }
  // ---- stage rel-pos table slice rows 313..711 (band index 0..398) ----
  for (int idx = tid; idx < TSR_ * DH_; idx += 128) {
    int r = idx >> 6, c = idx & 63;
    Ts[r * KS_ST + c] = (r < 399) ? (_Float16)relT[(313 + r) * DH_ + c] : (_Float16)0;
  }
  async_wait0();          // ASYNCcnt == 0 before the barrier publishes Ks
  __syncthreads();

  float* S = Sw + wid * 16 * S_ST;
  _Float16* P = Pw + wid * 16 * VT_ST;
  float* sI = sInv + wid * 16;

#pragma unroll 1
  for (int qt = wid; qt < 13; qt += 4) {
    // ---- Q A-fragments for both K-steps (dh=64), reused across all tiles ----
    const int arow = qt * 16 + col16;
    const bool aok = (arow < C_);
    const _Float16* qrow = q16 + ((size_t)(b * NP_ + m * C_ + arow)) * 512 + h * DH_;
    h8 z = {};
    h8 q0 = aok ? *(const h8*)(qrow + 16 * hf) : z;
    h8 q1 = aok ? *(const h8*)(qrow + 8 + 16 * hf) : z;
    h8 q2 = aok ? *(const h8*)(qrow + 32 + 16 * hf) : z;
    h8 q3 = aok ? *(const h8*)(qrow + 40 + 16 * hf) : z;
    v16h a0 = __builtin_shufflevector(q0, q1, 0, 1, 2, 3, 4, 5, 6, 7,
                                      8, 9, 10, 11, 12, 13, 14, 15);
    v16h a1 = __builtin_shufflevector(q2, q3, 0, 1, 2, 3, 4, 5, 6, 7,
                                      8, 9, 10, 11, 12, 13, 14, 15);

    // ---- scores: S[row][key] = q . k ----
#pragma unroll 1
    for (int nt = 0; nt < 14; ++nt) {
      const _Float16* kcol = Ks + (nt * 16 + col16) * KS_ST + 16 * hf;
      v16h b0 = frag2(kcol, kcol + 8);
      v16h b1 = frag2(kcol + 32, kcol + 40);
      v8f acc = {};
      acc = WMMA_F16(a0, b0, acc);
      acc = WMMA_F16(a1, b1, acc);
#pragma unroll
      for (int r = 0; r < 8; ++r)
        S[(r + 8 * hf) * S_ST + nt * 16 + col16] = acc[r];
    }

    // ---- rel-pos band: band[c][j] = q[c] . T[j], scatter S[c][c+199-j] += ----
#pragma unroll 1
    for (int bt = 0; bt < 14; ++bt) {
      const int colbase = (qt + bt) * 16;
      const _Float16* tcol = Ts + (colbase + col16) * KS_ST + 16 * hf;
      v16h b0 = frag2(tcol, tcol + 8);
      v16h b1 = frag2(tcol + 32, tcol + 40);
      v8f acc = {};
      acc = WMMA_F16(a0, b0, acc);
      acc = WMMA_F16(a1, b1, acc);
#pragma unroll
      for (int r = 0; r < 8; ++r) {
        int row = r + 8 * hf;
        int key = qt * 16 + row + 199 - (colbase + col16);
        if (key >= 0 && key < CP_) S[row * S_ST + key] += acc[r];
      }
    }

    // ---- scale + mask + softmax (2 lanes per row, ds_bpermute xor-16) ----
    const int srow = col16;
    float mx = -3.0e38f;
    for (int cI = hf; cI < CP_; cI += 2) {
      float v = S[srow * S_ST + cI] * 0.125f;
      bool maskOut = (cI >= C_) || (last && (cI >= NR_ || (qt * 16 + srow) >= NR_));
      if (maskOut) v = -3.0e38f;
      S[srow * S_ST + cI] = v;
      mx = fmaxf(mx, v);
    }
    mx = fmaxf(mx, __int_as_float(
        __builtin_amdgcn_ds_bpermute((lane ^ 16) << 2, __float_as_int(mx))));
    float ssum = 0.f;
    for (int cI = hf; cI < CP_; cI += 2) {
      float e = __expf(S[srow * S_ST + cI] - mx);
      P[srow * VT_ST + cI] = (_Float16)e;
      ssum += e;
    }
    ssum += __int_as_float(
        __builtin_amdgcn_ds_bpermute((lane ^ 16) << 2, __float_as_int(ssum)));
    if (hf == 0) sI[srow] = 1.0f / ssum;

    // ---- out = P @ V (K = 224 keys, transposed-V B operand), 1/sum on store ----
#pragma unroll 1
    for (int nt = 0; nt < 4; ++nt) {
      v8f acc = {};
#pragma unroll 1
      for (int k0 = 0; k0 < CP_; k0 += 32) {
        const _Float16* prow = P + col16 * VT_ST + k0;
        v16h pa = frag2(prow + 16 * hf, prow + 8 + 16 * hf);
        const _Float16* vrow = Vt + (nt * 16 + col16) * VT_ST + k0 + 16 * hf;
        v16h vb = frag2(vrow, vrow + 8);
        acc = WMMA_F16(pa, vb, acc);
      }
#pragma unroll
      for (int r = 0; r < 8; ++r) {
        int row = r + 8 * hf;
        int rowg = qt * 16 + row;
        if (rowg < C_) {
          float o = acc[r] * sI[row];
          ctx[((size_t)(b * NP_ + m * C_ + rowg)) * 512 + h * DH_ + nt * 16 + col16] =
              (_Float16)o;
        }
      }
    }
  }
}

// =====================================================================
// Kernel 3: out[b,t,:] = ctx[b,t,:] @ wout + bias
// A tile staged with async global->LDS b128 copies (pure f16 copy).
// =====================================================================
__global__ void __launch_bounds__(256)
gemm_out_kernel(const _Float16* __restrict__ Ctx, const _Float16* __restrict__ Wp,
                const float* __restrict__ bias, float* __restrict__ Out) {
  __shared__ _Float16 Xs[16 * XS_ST];
  const int tid = threadIdx.x;
  const int mt = blockIdx.x;
  {
    const unsigned ldsX = lds_addr_of(Xs);
    const _Float16* src = Ctx + (size_t)mt * 16 * 512;
    for (int idx = tid; idx < 16 * 64; idx += 256) {      // 16 rows x 64 chunks
      int r = idx >> 6, c8 = (idx & 63) * 8;
      async_copy_b128(ldsX + (unsigned)(r * XS_ST + c8) * 2,
                      src + (size_t)r * 512 + c8);
    }
  }
  async_wait0();
  __syncthreads();

  const int lane = tid & 31, wid = tid >> 5;
  const int col16 = lane & 15, hf = lane >> 4;
  const int nt = blockIdx.y * 8 + wid;
  const int col = nt * 16 + col16;
  const _Float16* xrow = Xs + col16 * XS_ST;
  v8f acc = {};
#pragma unroll 2
  for (int k0 = 0; k0 < 512; k0 += 32) {
    v16h a = frag2(xrow + k0 + 8 * hf, xrow + k0 + 16 + 8 * hf);
    v16h bf = *(const v16h*)(Wp + (((size_t)nt * 16 + (k0 >> 5)) * 32 + lane) * 16);
    acc = WMMA_F16(a, bf, acc);
  }
  const float bcol = bias[col];
#pragma unroll
  for (int r = 0; r < 8; ++r) {
    int mrow = mt * 16 + r + 8 * hf;
    int bb = mrow / NP_, tt = mrow - bb * NP_;
    if (tt < N_) Out[((size_t)bb * N_ + tt) * D_ + col] = acc[r] + bcol;
  }
}

// =====================================================================
extern "C" void kernel_launch(void* const* d_in, const int* in_sizes, int n_in,
                              void* d_out, int out_size, void* d_ws, size_t ws_size,
                              hipStream_t stream) {
  (void)in_sizes; (void)n_in; (void)out_size; (void)ws_size;
  const float* x      = (const float*)d_in[0];   // [16,3900,512]
  const float* wq     = (const float*)d_in[1];   // [512,512]
  const float* wkv    = (const float*)d_in[2];   // [512,1024]
  const float* wout_w = (const float*)d_in[3];   // [512,512]
  const float* wout_b = (const float*)d_in[4];   // [512]
  const float* relT   = (const float*)d_in[5];   // [1025,64]
  float* out = (float*)d_out;

  // workspace (~190 MB): packed f16 weights + q16 + kv16. ctx aliases q16
  // (each attention block reads its q tile before overwriting the same
  //  rows/cols; all other blocks touch disjoint row x col regions).
  char* ws = (char*)d_ws;
  _Float16* wq16  = (_Float16*)ws; ws += (size_t)512 * 512 * 2;
  _Float16* wkv16 = (_Float16*)ws; ws += (size_t)512 * 1024 * 2;
  _Float16* wo16  = (_Float16*)ws; ws += (size_t)512 * 512 * 2;
  _Float16* q16   = (_Float16*)ws; ws += (size_t)B_ * NP_ * 512 * 2;
  _Float16* kv16  = (_Float16*)ws; ws += (size_t)B_ * NP_ * 1024 * 2;
  _Float16* ctx16 = q16;

  pack_w_kernel<<<1024, 256, 0, stream>>>(wq, wq16, 512);
  pack_w_kernel<<<2048, 256, 0, stream>>>(wkv, wkv16, 1024);
  pack_w_kernel<<<1024, 256, 0, stream>>>(wout_w, wo16, 512);

  gemm_qkv_kernel<<<dim3(4000, 4), 256, 0, stream>>>(x, wq16, q16, 512);
  gemm_qkv_kernel<<<dim3(4000, 8), 256, 0, stream>>>(x, wkv16, kv16, 1024);

  size_t smem = (size_t)(CP_ * KS_ST + DH_ * VT_ST + TSR_ * KS_ST) * sizeof(_Float16)
              + (size_t)4 * 16 * S_ST * sizeof(float)        // Sw
              + (size_t)4 * 16 * VT_ST * sizeof(_Float16)    // Pw
              + (size_t)4 * 16 * sizeof(float);              // sInv
  attn_kernel<<<dim3(NB_ * H_, B_), 128, smem, stream>>>(q16, kv16, relT, ctx16);

  gemm_out_kernel<<<dim3(4000, 4), 256, 0, stream>>>(ctx16, wo16, wout_b, out);
}
